// CausalSelfAttention_32444182954579
// MI455X (gfx1250) — matrix operand order repaired and checked by reading, not compiled, not run on hardware
//
#include <hip/hip_runtime.h>

typedef __attribute__((ext_vector_type(16))) _Float16 v16h;
typedef __attribute__((ext_vector_type(8)))  _Float16 v8h;
typedef __attribute__((ext_vector_type(8)))  float    v8f;
typedef __attribute__((ext_vector_type(4)))  int      v4i;

#define B_SZ    2
#define S_LEN   2048
#define DMODEL  1024
#define NHEADS  16
#define HDIM    64
#define NROWS   (B_SZ * S_LEN)   /* 4096 */
#define SM_SCALE 0.125f          /* 1/sqrt(64) */

#define WMMA_F32_F16(a, b, c) \
  __builtin_amdgcn_wmma_f32_16x16x32_f16(false, (a), false, (b), (short)0, (c), false, false)

// ---- CDNA5 async global->LDS path (ASYNCcnt), with safe fallback ----------
#if __has_builtin(__builtin_amdgcn_global_load_async_to_lds_b128)
#define HAVE_ASYNC 1
#else
#define HAVE_ASYNC 0
#endif

#if HAVE_ASYNC
#if __has_builtin(__builtin_amdgcn_s_wait_asynccnt)
#define WAIT_ASYNC(n) __builtin_amdgcn_s_wait_asynccnt(n)
#else
#define WAIT_ASYNC(n) asm volatile("s_wait_asynccnt " #n ::: "memory")
#endif
typedef __attribute__((address_space(1))) v4i gv4i;
typedef __attribute__((address_space(3))) v4i lv4i;
__device__ __forceinline__ void async_cp16(const void* g, void* l) {
  __builtin_amdgcn_global_load_async_to_lds_b128((gv4i*)g, (lv4i*)l, 0, 0);
}
#else
#define WAIT_ASYNC(n) ((void)0)
__device__ __forceinline__ void async_cp16(const void* g, void* l) {
  *(v8h*)l = *(const v8h*)g;
}
#endif

// ---------------------------------------------------------------------------
// Fragment loaders (wave32).
// A 16x32 f16 layout: lane<16 holds row=lane, K {0..7,16..23}; lane>=16 holds
// row=lane-16, K {8..15,24..31}  -> two contiguous 16B chunks per lane.
// B 32x16 f16 layout: lane gives N=lane&15; lanes<16 hold K {0..7,16..23},
// lanes>=16 hold K {8..15,24..31} -> two contiguous 16B chunks (from [N][K]).
// ---------------------------------------------------------------------------
__device__ __forceinline__ v16h frag_a(const _Float16* base, int ldh) {
  const int lane = threadIdx.x & 31;
  const _Float16* p = base + (size_t)(lane & 15) * ldh + ((lane >> 4) << 3);
  v8h lo = *(const v8h*)(p);
  v8h hi = *(const v8h*)(p + 16);
  return __builtin_shufflevector(lo, hi, 0, 1, 2, 3, 4, 5, 6, 7,
                                 8, 9, 10, 11, 12, 13, 14, 15);
}

__device__ __forceinline__ v16h frag_b(const _Float16* base, int ldh) {
  const int lane = threadIdx.x & 31;
  const _Float16* p = base + (size_t)(lane & 15) * ldh + ((lane >> 4) << 3);
  v8h lo = *(const v8h*)(p);
  v8h hi = *(const v8h*)(p + 16);
  return __builtin_shufflevector(lo, hi, 0, 1, 2, 3, 4, 5, 6, 7,
                                 8, 9, 10, 11, 12, 13, 14, 15);
}

// ---------------------------------------------------------------------------
// GEMM: C[4096,1024] = A[4096,1024] * B[1024,1024]; fp32 global, fp16 LDS,
// fp32 WMMA accumulate. Block tile 128x64, BK=64, 8 waves (4x2), 2x2 WMMA per
// wave per K-chunk. Register double-buffered global->LDS pipeline.
// ---------------------------------------------------------------------------
__global__ void __launch_bounds__(256) gemm_f32_wmma(
    const float* __restrict__ A, const float* __restrict__ Bm,
    float* __restrict__ C) {
  constexpr int K = DMODEL, N = DMODEL, BK = 64;
  __shared__ _Float16 As[128][72];  // [M][K] pad->72 halves (144B rows)
  __shared__ _Float16 Bs[64][72];   // [N][K] (transposed on store)

  const int tid = threadIdx.x;
  const int w = tid >> 5, lane = tid & 31;
  const int wm = w >> 1, wn = w & 1;           // 4x2 wave grid
  const int m0 = blockIdx.y * 128, n0 = blockIdx.x * 64;

  float4 ar[8], br[4];
  v8f acc[2][2] = {};

  // stage 0: load first tile into registers
#pragma unroll
  for (int i = 0; i < 8; ++i) {
    const int f4 = tid + i * 256;                  // A: 128x64 -> 2048 float4
    ar[i] = *(const float4*)(A + (size_t)(m0 + (f4 >> 4)) * K + ((f4 & 15) << 2));
  }
#pragma unroll
  for (int i = 0; i < 4; ++i) {
    const int f4 = tid + i * 256;                  // B: 64x64 -> 1024 float4
    br[i] = *(const float4*)(Bm + (size_t)(f4 >> 4) * N + n0 + ((f4 & 15) << 2));
  }

  for (int k0 = 0; k0 < K; k0 += BK) {
    // convert + store the staged tile to LDS
#pragma unroll
    for (int i = 0; i < 8; ++i) {
      const int f4 = tid + i * 256;
      const int r = f4 >> 4, kc = (f4 & 15) << 2;
      As[r][kc + 0] = (_Float16)ar[i].x;
      As[r][kc + 1] = (_Float16)ar[i].y;
      As[r][kc + 2] = (_Float16)ar[i].z;
      As[r][kc + 3] = (_Float16)ar[i].w;
    }
#pragma unroll
    for (int i = 0; i < 4; ++i) {
      const int f4 = tid + i * 256;
      const int r = f4 >> 4, nc = (f4 & 15) << 2;
      Bs[nc + 0][r] = (_Float16)br[i].x;
      Bs[nc + 1][r] = (_Float16)br[i].y;
      Bs[nc + 2][r] = (_Float16)br[i].z;
      Bs[nc + 3][r] = (_Float16)br[i].w;
    }
    __syncthreads();

    // prefetch next tile into registers while WMMAs run
    if (k0 + BK < K) {
#pragma unroll
      for (int i = 0; i < 8; ++i) {
        const int f4 = tid + i * 256;
        ar[i] = *(const float4*)(A + (size_t)(m0 + (f4 >> 4)) * K + k0 + BK +
                                 ((f4 & 15) << 2));
      }
#pragma unroll
      for (int i = 0; i < 4; ++i) {
        const int f4 = tid + i * 256;
        br[i] = *(const float4*)(Bm + (size_t)(k0 + BK + (f4 >> 4)) * N + n0 +
                                 ((f4 & 15) << 2));
      }
    }

#pragma unroll
    for (int c = 0; c < 2; ++c) {                  // two K=32 chunks
      const v16h a0 = frag_a(&As[wm * 32][c * 32], 72);
      const v16h a1 = frag_a(&As[wm * 32 + 16][c * 32], 72);
      const v16h b0 = frag_b(&Bs[wn * 32][c * 32], 72);
      const v16h b1 = frag_b(&Bs[wn * 32 + 16][c * 32], 72);
      acc[0][0] = WMMA_F32_F16(a0, b0, acc[0][0]);
      acc[0][1] = WMMA_F32_F16(a0, b1, acc[0][1]);
      acc[1][0] = WMMA_F32_F16(a1, b0, acc[1][0]);
      acc[1][1] = WMMA_F32_F16(a1, b1, acc[1][1]);
    }
    __syncthreads();
  }

  // C/D layout: lane<16 -> N=lane, M=v; lane>=16 -> N=lane-16, M=v+8
  const int rw = (lane >> 4) << 3, col = lane & 15;
#pragma unroll
  for (int mi = 0; mi < 2; ++mi)
#pragma unroll
    for (int ni = 0; ni < 2; ++ni)
#pragma unroll
      for (int v = 0; v < 8; ++v) {
        const int r = m0 + wm * 32 + mi * 16 + rw + v;
        const int c = n0 + wn * 32 + ni * 16 + col;
        C[(size_t)r * N + c] = acc[mi][ni][v];
      }
}

// ---------------------------------------------------------------------------
// QK RMS-norm + RoPE + layout conversion. One wave per (b,s,h); lane t owns
// the RoPE pair (t, t+32). Outputs Qh/Kh fp16 [B,H,S,64]; Vt fp16 [B,H,64,S].
// ---------------------------------------------------------------------------
__global__ void __launch_bounds__(32) qkv_post(
    const float* __restrict__ Qf, const float* __restrict__ Kf,
    const float* __restrict__ Vf, _Float16* __restrict__ Qh,
    _Float16* __restrict__ Kh, _Float16* __restrict__ Vt) {
  const int h = blockIdx.x;
  const int row = blockIdx.y;             // b*S + s
  const int s = row & (S_LEN - 1);
  const int b = row >> 11;                // row / 2048
  const int t = threadIdx.x;              // 0..31

  const size_t base = (size_t)row * DMODEL + h * HDIM;
  float q1 = Qf[base + t], q2 = Qf[base + t + 32];
  float k1 = Kf[base + t], k2 = Kf[base + t + 32];
  const float v1 = Vf[base + t], v2 = Vf[base + t + 32];

  float sq = q1 * q1 + q2 * q2;
  float sk = k1 * k1 + k2 * k2;
#pragma unroll
  for (int off = 16; off > 0; off >>= 1) {
    sq += __shfl_xor(sq, off);
    sk += __shfl_xor(sk, off);
  }
  const float rq = rsqrtf(sq * (1.0f / HDIM) + 1e-6f);
  const float rk = rsqrtf(sk * (1.0f / HDIM) + 1e-6f);
  q1 *= rq; q2 *= rq; k1 *= rk; k2 *= rk;

  // inv_freq = theta^(-t/32) = exp(-ln(1e4) * t/32)
  const float ang = (float)s * __expf(-9.2103403719761836f * ((float)t * (1.0f / 32.0f)));
  const float cc = __cosf(ang), sn = __sinf(ang);
  const float qo1 = q1 * cc - q2 * sn, qo2 = q2 * cc + q1 * sn;
  const float ko1 = k1 * cc - k2 * sn, ko2 = k2 * cc + k1 * sn;

  const size_t bh = (size_t)(b * NHEADS + h);
  const size_t qb = (bh * S_LEN + s) * HDIM;
  Qh[qb + t] = (_Float16)qo1;
  Qh[qb + t + 32] = (_Float16)qo2;
  Kh[qb + t] = (_Float16)ko1;
  Kh[qb + t + 32] = (_Float16)ko2;
  Vt[(bh * HDIM + t) * S_LEN + s] = (_Float16)v1;
  Vt[(bh * HDIM + t + 32) * S_LEN + s] = (_Float16)v2;
}

// ---------------------------------------------------------------------------
// Flash attention (causal). 8 waves x 16 q-rows = 128 q-rows per block; key
// tiles of 32 double-buffered in LDS via async global->LDS DMA (ASYNCcnt).
// ---------------------------------------------------------------------------
__global__ void __launch_bounds__(256) attn_fwd(
    const _Float16* __restrict__ Qh, const _Float16* __restrict__ Kh,
    const _Float16* __restrict__ Vt, float* __restrict__ O) {
  const int bh = blockIdx.y;
  const int b = bh >> 4, h = bh & 15;
  const int qblk = blockIdx.x;
  const int tid = threadIdx.x, w = tid >> 5, lane = tid & 31;
  const int qbase = qblk * 128 + w * 16;

  __shared__ _Float16 Ks[2][32][72];  // [buf][key][hd] : B-frags for Q*K^T
  __shared__ _Float16 Vs[2][64][40];  // [buf][hd][key] : B-frags for P*V
  __shared__ float    Sc[8][16][33];  // per-wave fp32 scores
  __shared__ _Float16 Ph[8][16][40];  // per-wave fp16 probs (A layout)
  __shared__ float    Mrow[8][16], Lrow[8][16], Rsc[8][16];

  // Q fragments live in registers for the whole kernel (A 16x32 layout).
  const _Float16* Qp = Qh + ((size_t)bh * S_LEN + qbase) * HDIM;
  const v16h qa0 = frag_a(Qp, HDIM);
  const v16h qa1 = frag_a(Qp + 32, HDIM);

  if (lane < 16) { Mrow[w][lane] = -1e30f; Lrow[w][lane] = 0.0f; }
  v8f o0 = {}, o1 = {}, o2 = {}, o3 = {};

  const int kr = tid >> 3, kcl = (tid & 7) << 3;   // K-tile copy slot
  const int vr = tid >> 2, vcl = (tid & 3) << 3;   // V-tile copy slot
  const _Float16* Kbase = Kh + (size_t)bh * S_LEN * HDIM;
  const _Float16* Vbase = Vt + (size_t)bh * HDIM * S_LEN;

  auto issue_tile = [&](int k0, int bufi) {
    async_cp16(Kbase + (size_t)(k0 + kr) * HDIM + kcl, &Ks[bufi][kr][kcl]);
    async_cp16(Vbase + (size_t)vr * S_LEN + k0 + vcl, &Vs[bufi][vr][vcl]);
  };

  const int r = lane & 15, hc = lane >> 4;   // softmax ownership
  const int kend = qblk * 128 + 128;         // block-uniform causal bound

  issue_tile(0, 0);
  int buf = 0;
  for (int k0 = 0; k0 < kend; k0 += 32, buf ^= 1) {
    const bool more = (k0 + 32 < kend);      // uniform
    if (more) {
      issue_tile(k0 + 32, buf ^ 1);          // buf^1 fully consumed last iter
      WAIT_ASYNC(2);                         // this tile's 2 copies done
    } else {
      WAIT_ASYNC(0);
    }
    __syncthreads();

    // S = Q * K^T : two 16-wide key column blocks, hd split into 2 K-chunks
    v8f s0 = {}, s1 = {};
    {
      const v16h kb00 = frag_b(&Ks[buf][0][0], 72);
      const v16h kb01 = frag_b(&Ks[buf][0][32], 72);
      const v16h kb10 = frag_b(&Ks[buf][16][0], 72);
      const v16h kb11 = frag_b(&Ks[buf][16][32], 72);
      s0 = WMMA_F32_F16(qa0, kb00, s0);
      s0 = WMMA_F32_F16(qa1, kb01, s0);
      s1 = WMMA_F32_F16(qa0, kb10, s1);
      s1 = WMMA_F32_F16(qa1, kb11, s1);
    }
    {
      const int rw = hc << 3;
#pragma unroll
      for (int v = 0; v < 8; ++v) {
        Sc[w][rw + v][r] = s0[v];
        Sc[w][rw + v][16 + r] = s1[v];
      }
    }
    __syncthreads();

    // Online softmax: lane owns (row=r, cols hc*16..hc*16+15)
    const int qg = qbase + r;
    float mloc = -1e30f;
#pragma unroll
    for (int j = 0; j < 16; ++j) {
      const int key = k0 + hc * 16 + j;
      float sv = Sc[w][r][hc * 16 + j] * SM_SCALE;
      sv = (key <= qg) ? sv : -1e30f;
      Sc[w][r][hc * 16 + j] = sv;
      mloc = fmaxf(mloc, sv);
    }
    mloc = fmaxf(mloc, __shfl_xor(mloc, 16));
    const float mold = Mrow[w][r];
    const float mnew = fmaxf(mold, mloc);
    float psum = 0.0f;
#pragma unroll
    for (int j = 0; j < 16; ++j) {
      const float sv = Sc[w][r][hc * 16 + j];
      const float p = (sv < -1e29f) ? 0.0f : __expf(sv - mnew);
      Ph[w][r][hc * 16 + j] = (_Float16)p;
      psum += p;
    }
    psum += __shfl_xor(psum, 16);
    const float resc = __expf(mold - mnew);
    if (lane < 16) {
      Mrow[w][r] = mnew;
      Lrow[w][r] = Lrow[w][r] * resc + psum;
      Rsc[w][r] = resc;
    }
    __syncthreads();

    // Rescale accumulators and O += P * V
    {
      const int rw = hc << 3;
      float rs[8];
#pragma unroll
      for (int v = 0; v < 8; ++v) rs[v] = Rsc[w][rw + v];
#pragma unroll
      for (int v = 0; v < 8; ++v) {
        o0[v] *= rs[v]; o1[v] *= rs[v]; o2[v] *= rs[v]; o3[v] *= rs[v];
      }
      const v16h pa = frag_a(&Ph[w][0][0], 40);
      const v16h vb0 = frag_b(&Vs[buf][0][0], 40);
      const v16h vb1 = frag_b(&Vs[buf][16][0], 40);
      const v16h vb2 = frag_b(&Vs[buf][32][0], 40);
      const v16h vb3 = frag_b(&Vs[buf][48][0], 40);
      o0 = WMMA_F32_F16(pa, vb0, o0);
      o1 = WMMA_F32_F16(pa, vb1, o1);
      o2 = WMMA_F32_F16(pa, vb2, o2);
      o3 = WMMA_F32_F16(pa, vb3, o3);
    }
    __syncthreads();  // all waves done with buf -> safe to refill next iter
  }

  // Normalize by l and write fp32 attn output in [B,S,H*64] layout
  const int rw = hc << 3, col = lane & 15;
#pragma unroll
  for (int v = 0; v < 8; ++v) {
    const float inv = 1.0f / Lrow[w][rw + v];
    const int q = qbase + rw + v;
    float* op = O + ((size_t)b * S_LEN + q) * DMODEL + h * HDIM + col;
    op[0]  = o0[v] * inv;
    op[16] = o1[v] * inv;
    op[32] = o2[v] * inv;
    op[48] = o3[v] * inv;
  }
}

// ---------------------------------------------------------------------------
extern "C" void kernel_launch(void* const* d_in, const int* in_sizes, int n_in,
                              void* d_out, int out_size, void* d_ws,
                              size_t ws_size, hipStream_t stream) {
  (void)in_sizes; (void)n_in; (void)out_size; (void)ws_size;
  const float* x   = (const float*)d_in[0];
  const float* w_q = (const float*)d_in[1];
  const float* w_k = (const float*)d_in[2];
  const float* w_v = (const float*)d_in[3];
  const float* w_o = (const float*)d_in[4];
  // d_in[5] = causal mask: implemented analytically in attn_fwd.

  const size_t NE = (size_t)NROWS * DMODEL;  // 4Mi elements
  float* Qf = (float*)d_ws;                  // fp32 Q projection
  float* Kf = Qf + NE;
  float* Vf = Kf + NE;
  float* At = Vf + NE;                       // fp32 attention output
  _Float16* Qhh = (_Float16*)(At + NE);      // fp16 [B,H,S,64]
  _Float16* Khh = Qhh + NE;
  _Float16* Vtt = Khh + NE;                  // fp16 [B,H,64,S]

  const dim3 gg(DMODEL / 64, NROWS / 128);   // (16, 32)
  gemm_f32_wmma<<<gg, 256, 0, stream>>>(x, w_q, Qf);
  gemm_f32_wmma<<<gg, 256, 0, stream>>>(x, w_k, Kf);
  gemm_f32_wmma<<<gg, 256, 0, stream>>>(x, w_v, Vf);
  qkv_post<<<dim3(NHEADS, NROWS), 32, 0, stream>>>(Qf, Kf, Vf, Qhh, Khh, Vtt);
  attn_fwd<<<dim3(S_LEN / 128, B_SZ * NHEADS), 256, 0, stream>>>(Qhh, Khh, Vtt, At);
  gemm_f32_wmma<<<gg, 256, 0, stream>>>(At, w_o, (float*)d_out);
}
